// Seq2SeqCodeBERT_86672440033544
// MI455X (gfx1250) — matrix operand order repaired and checked by reading, not compiled, not run on hardware
//
#include <hip/hip_runtime.h>
#include <hip/hip_bf16.h>
#include <math.h>

// ---------------- model constants ----------------
#define H      768
#define NHEAD  12
#define DH     64
#define FFD    3072
#define VOC    50265
#define NENC   2
#define NDEC   2
#define BB     2          // batch
#define SS     128        // encoder seq len
#define BEAM   4
#define MAXLEN 8
#define LB     (MAXLEN+1) // decoder seq len
#define RENC   (BB*SS)    // 256 encoder rows
#define RB     (BB*BEAM)  // 8 beam rows
#define RDEC   (RB*LB)    // 72 decoder rows
#define EOS_TOK 2

#define ACT_NONE 0
#define ACT_GELU 1
#define ACT_RELU 2
#define ACT_TANH 3

typedef __attribute__((ext_vector_type(16))) _Float16 v16h;
typedef __attribute__((ext_vector_type(8)))  _Float16 v8h;
typedef __attribute__((ext_vector_type(4)))  _Float16 h4;
typedef __attribute__((ext_vector_type(8)))  float    v8f;

// ---------------- WMMA GEMM:  C = act(A[MxK] * B[KxN] + bias) ----------------
// 128 threads = 4 waves; block tile 32(M) x 64(N); each wave owns 16x32 (2 WMMAs
// sharing one A fragment). LDS: As[m][k], Bs[n][k] (B transposed in LDS so both
// fragments are contiguous ds_load_b128 runs). Register double-buffered K loop.
// Requires K % 32 == 0 (all call sites use K in {768, 3072}).
__device__ __forceinline__ float activate(float v, int act) {
  if (act == ACT_GELU) {
    float x3 = v * v * v;
    return 0.5f * v * (1.0f + tanhf(0.7978845608028654f * (v + 0.044715f * x3)));
  } else if (act == ACT_RELU) {
    return v > 0.0f ? v : 0.0f;
  } else if (act == ACT_TANH) {
    return tanhf(v);
  }
  return v;
}

__device__ __forceinline__ float f4c(const float4& v, int p) {
  const float a[4] = {v.x, v.y, v.z, v.w};
  return a[p];
}

__global__ __launch_bounds__(128)
void k_gemm_wmma(const float* __restrict__ A, const float* __restrict__ Bm,
                 const float* __restrict__ bias, float* __restrict__ C,
                 int M, int N, int K, int transB, int act) {
  __shared__ __align__(16) _Float16 As[32][40];  // [m][k], row stride 80 B
  __shared__ __align__(16) _Float16 Bs[64][40];  // [n][k], row stride 80 B
  const int tid  = threadIdx.x;
  const int lane = tid & 31;
  const int wave = tid >> 5;
  const int m0 = blockIdx.y * 32, n0 = blockIdx.x * 64;
  const int wr = (wave >> 1) << 4;       // wave row offset 0/16
  const int wn = (wave & 1) << 5;        // wave col offset 0/32
  const int l16 = lane & 15;
  const int hiA = (lane < 16) ? 0 : 16;  // A half-wave K offset
  const int hiB = (lane < 16) ? 0 : 16;  // B half-wave K offset

  const float4 f4z = make_float4(0.f, 0.f, 0.f, 0.f);
  float4 ra[2], rb[4];

  // fetch one 32-wide K slab into registers (global_load_b128 chunks)
  auto fetch = [&](int kk) {
#pragma unroll
    for (int u = 0; u < 2; ++u) {                  // A: 32 rows x 8 float4
      int i = tid + u * 128, r = i >> 3, c4 = (i & 7) << 2;
      int gm = m0 + r;
      ra[u] = (gm < M) ? *(const float4*)(A + (size_t)gm * K + kk + c4) : f4z;
    }
    if (transB) {                                  // B global [N,K]: rows of K
#pragma unroll
      for (int u = 0; u < 4; ++u) {                // 64 n-rows x 8 float4
        int i = tid + u * 128, n = i >> 3, c4 = (i & 7) << 2;
        int gn = n0 + n;
        rb[u] = (gn < N) ? *(const float4*)(Bm + (size_t)gn * K + kk + c4) : f4z;
      }
    } else {                                       // B global [K,N]: 4x4 micro-tile
      int kb = tid >> 4, nb = tid & 15;            // k rows kb*4.., n cols nb*4..
      int gn = n0 + (nb << 2);
#pragma unroll
      for (int q = 0; q < 4; ++q)
        rb[q] = (gn < N) ? *(const float4*)(Bm + (size_t)(kk + (kb << 2) + q) * N + gn) : f4z;
    }
  };

  // convert f32->f16 and stage registers into LDS (ds_store_b64 chunks)
  auto stage = [&]() {
#pragma unroll
    for (int u = 0; u < 2; ++u) {
      int i = tid + u * 128, r = i >> 3, c4 = (i & 7) << 2;
      h4 h = {(_Float16)ra[u].x, (_Float16)ra[u].y, (_Float16)ra[u].z, (_Float16)ra[u].w};
      *(h4*)&As[r][c4] = h;
    }
    if (transB) {
#pragma unroll
      for (int u = 0; u < 4; ++u) {
        int i = tid + u * 128, n = i >> 3, c4 = (i & 7) << 2;
        h4 h = {(_Float16)rb[u].x, (_Float16)rb[u].y, (_Float16)rb[u].z, (_Float16)rb[u].w};
        *(h4*)&Bs[n][c4] = h;
      }
    } else {
      int kb = tid >> 4, nb = tid & 15;
#pragma unroll
      for (int p = 0; p < 4; ++p) {                // transpose in registers
        h4 h = {(_Float16)f4c(rb[0], p), (_Float16)f4c(rb[1], p),
                (_Float16)f4c(rb[2], p), (_Float16)f4c(rb[3], p)};
        *(h4*)&Bs[(nb << 2) + p][kb << 2] = h;
      }
    }
  };

  v8f acc0 = {0.f, 0.f, 0.f, 0.f, 0.f, 0.f, 0.f, 0.f};
  v8f acc1 = {0.f, 0.f, 0.f, 0.f, 0.f, 0.f, 0.f, 0.f};

  fetch(0);
  stage();
  __syncthreads();
  for (int kk = 0; kk < K; kk += 32) {
    const bool hasNext = (kk + 32) < K;
    if (hasNext) fetch(kk + 32);                   // overlap with WMMA below

    // fragments: contiguous 16B LDS runs -> ds_load_b128
    const v8h a0 = *(const v8h*)&As[wr + l16][hiA];
    const v8h a1 = *(const v8h*)&As[wr + l16][hiA + 8];
    v16h af = __builtin_shufflevector(a0, a1, 0,1,2,3,4,5,6,7,8,9,10,11,12,13,14,15);
    const v8h b00 = *(const v8h*)&Bs[wn + l16][hiB];
    const v8h b01 = *(const v8h*)&Bs[wn + l16][hiB + 8];
    v16h bf0 = __builtin_shufflevector(b00, b01, 0,1,2,3,4,5,6,7,8,9,10,11,12,13,14,15);
    const v8h b10 = *(const v8h*)&Bs[wn + 16 + l16][hiB];
    const v8h b11 = *(const v8h*)&Bs[wn + 16 + l16][hiB + 8];
    v16h bf1 = __builtin_shufflevector(b10, b11, 0,1,2,3,4,5,6,7,8,9,10,11,12,13,14,15);

    acc0 = __builtin_amdgcn_wmma_f32_16x16x32_f16(false, af, false, bf0,
                                                  (short)0, acc0, false, false);
    acc1 = __builtin_amdgcn_wmma_f32_16x16x32_f16(false, af, false, bf1,
                                                  (short)0, acc1, false, false);
    __syncthreads();
    if (hasNext) stage();
    __syncthreads();
  }

  // C/D layout: VGPR r -> row r (lanes 0-15) / r+8 (lanes 16-31), col = lane%16
#pragma unroll
  for (int r = 0; r < 8; ++r) {
    int m = m0 + wr + r + ((lane < 16) ? 0 : 8);
    if (m < M) {
      int n1 = n0 + wn + l16;
      if (n1 < N) {
        float v = acc0[r];
        if (bias) v += bias[n1];
        C[(size_t)m * N + n1] = activate(v, act);
      }
      int n2 = n1 + 16;
      if (n2 < N) {
        float v = acc1[r];
        if (bias) v += bias[n2];
        C[(size_t)m * N + n2] = activate(v, act);
      }
    }
  }
}

// ---------------- fused (x + h) LayerNorm, in place on x; H = 768 ----------------
__global__ __launch_bounds__(256)
void k_add_ln(float* __restrict__ x, const float* __restrict__ h,
              const float* __restrict__ g, const float* __restrict__ b) {
  __shared__ float red[256];
  const int row = blockIdx.x, tid = threadIdx.x;
  float v[3];
#pragma unroll
  for (int j = 0; j < 3; ++j) {
    int d = j * 256 + tid;
    float t = x[(size_t)row * H + d];
    if (h) t += h[(size_t)row * H + d];
    v[j] = t;
  }
  red[tid] = v[0] + v[1] + v[2];
  __syncthreads();
  for (int o = 128; o > 0; o >>= 1) { if (tid < o) red[tid] += red[tid + o]; __syncthreads(); }
  float mean = red[0] * (1.0f / (float)H);
  __syncthreads();
  float sq = 0.0f;
#pragma unroll
  for (int j = 0; j < 3; ++j) { float d = v[j] - mean; sq += d * d; }
  red[tid] = sq;
  __syncthreads();
  for (int o = 128; o > 0; o >>= 1) { if (tid < o) red[tid] += red[tid + o]; __syncthreads(); }
  float inv = rsqrtf(red[0] * (1.0f / (float)H) + 1e-5f);
#pragma unroll
  for (int j = 0; j < 3; ++j) {
    int d = j * 256 + tid;
    x[(size_t)row * H + d] = (v[j] - mean) * inv * g[d] + b[d];
  }
}

// ---------------- softmax attention: one block per (batch, head, query) ----------------
// causal=1 -> bias = (s<=t ? 0 : -1e4); else bias = biasBK[(b/bias_div)*Skv + s]
// KV batch index = b / kv_div (beam broadcast of encoder memory)
__global__ __launch_bounds__(256)
void k_attn(const float* __restrict__ Q, const float* __restrict__ Kt,
            const float* __restrict__ Vt, float* __restrict__ O,
            const float* __restrict__ biasBK, int bias_div, int causal,
            int T, int Skv, int kv_div, float scale) {
  __shared__ float sc[SS];
  __shared__ float red[256];
  __shared__ float qv[DH];
  const int t = blockIdx.x, hh = blockIdx.y, b = blockIdx.z;
  const int tid = threadIdx.x;
  const int kb = b / kv_div;
  const float* q = Q + ((size_t)(b * T + t)) * H + hh * DH;
  if (tid < DH) qv[tid] = q[tid];
  __syncthreads();

  for (int s = tid; s < Skv; s += 256) {
    const float* kp = Kt + ((size_t)(kb * Skv + s)) * H + hh * DH;
    float a = 0.0f;
#pragma unroll 8
    for (int d = 0; d < DH; ++d) a += qv[d] * kp[d];
    float bi = causal ? ((s <= t) ? 0.0f : -10000.0f)
                      : biasBK[(b / bias_div) * Skv + s];
    sc[s] = a * scale + bi;
  }
  __syncthreads();
  float m = -3.0e38f;
  for (int s = tid; s < Skv; s += 256) m = fmaxf(m, sc[s]);
  red[tid] = m;
  __syncthreads();
  for (int o = 128; o > 0; o >>= 1) { if (tid < o) red[tid] = fmaxf(red[tid], red[tid + o]); __syncthreads(); }
  float mx = red[0];
  __syncthreads();
  float su = 0.0f;
  for (int s = tid; s < Skv; s += 256) { float e = __expf(sc[s] - mx); sc[s] = e; su += e; }
  red[tid] = su;
  __syncthreads();
  for (int o = 128; o > 0; o >>= 1) { if (tid < o) red[tid] += red[tid + o]; __syncthreads(); }
  float denom = red[0];
  __syncthreads();
  if (tid < DH) {
    float o = 0.0f;
    for (int s = 0; s < Skv; ++s)
      o += sc[s] * Vt[((size_t)(kb * Skv + s)) * H + hh * DH + tid];
    O[((size_t)(b * T + t)) * H + hh * DH + tid] = o / denom;
  }
}

// ---------------- embedding (word + pos); LN done by k_add_ln(h=null) ----------------
__global__ void k_embed(const int* __restrict__ ids, const float* __restrict__ wemb,
                        const float* __restrict__ pemb, float* __restrict__ out,
                        int R, int L) {
  int i = blockIdx.x * 256 + threadIdx.x;
  if (i >= R * H) return;
  int r = i / H, d = i - r * H;
  out[i] = wemb[(size_t)ids[r] * H + d] + pemb[(size_t)(r % L) * H + d];
}

__global__ void k_encbias(const int* __restrict__ mask, float* __restrict__ bias) {
  int i = blockIdx.x * 256 + threadIdx.x;
  if (i < BB * SS) bias[i] = (1.0f - (float)mask[i]) * -1e9f;
}

__global__ void k_extract(const float* __restrict__ x, float* __restrict__ out, int t) {
  int i = blockIdx.x * 256 + threadIdx.x;
  if (i >= RB * H) return;
  int rb = i / H, d = i - rb * H;
  out[i] = x[((size_t)(rb * LB + t)) * H + d];
}

__global__ void k_init_state(int* __restrict__ tok, float* __restrict__ scores,
                             float* __restrict__ fin_s, int* __restrict__ fin_t) {
  int i = threadIdx.x;
  if (i < RDEC) tok[i] = 0;                       // SOS == 0, rest zero
  if (i < RB) { scores[i] = ((i % BEAM) == 0) ? 0.0f : -1e20f; fin_s[i] = -1e30f; }
  if (i < BB * BEAM * MAXLEN) fin_t[i] = 0;
}

// ---------------- row-wise max + log-sum-exp over vocab ----------------
__global__ __launch_bounds__(256)
void k_row_lse(const float* __restrict__ L, float* __restrict__ mx,
               float* __restrict__ lse, int N) {
  __shared__ float red[256];
  const int r = blockIdx.x, tid = threadIdx.x;
  const float* p = L + (size_t)r * N;
  float m = -3.0e38f;
  for (int i = tid; i < N; i += 256) m = fmaxf(m, p[i]);
  red[tid] = m;
  __syncthreads();
  for (int o = 128; o > 0; o >>= 1) { if (tid < o) red[tid] = fmaxf(red[tid], red[tid + o]); __syncthreads(); }
  float rm = red[0];
  __syncthreads();
  float s = 0.0f;
  for (int i = tid; i < N; i += 256) s += __expf(p[i] - rm);
  red[tid] = s;
  __syncthreads();
  for (int o = 128; o > 0; o >>= 1) { if (tid < o) red[tid] += red[tid + o]; __syncthreads(); }
  if (tid == 0) { mx[r] = rm; lse[r] = logf(red[0]); }
}

// ---------------- top-4 over beam*V per batch (stable: lowest flat index wins ties) ----
__device__ __forceinline__ void top4_insert(float* tv, int* ti, float val, int idx) {
  if (val > tv[3] || (val == tv[3] && idx < ti[3])) {
    int pos = 3;
    while (pos > 0 && (val > tv[pos - 1] || (val == tv[pos - 1] && idx < ti[pos - 1]))) {
      tv[pos] = tv[pos - 1]; ti[pos] = ti[pos - 1]; --pos;
    }
    tv[pos] = val; ti[pos] = idx;
  }
}

__global__ __launch_bounds__(256)
void k_topk(const float* __restrict__ logits, const float* __restrict__ mx,
            const float* __restrict__ lse, const float* __restrict__ scores,
            const int* __restrict__ tin, int t,
            float* __restrict__ best, int* __restrict__ bidx) {
  __shared__ float lv[1024];
  __shared__ int   li[1024];
  const int b = blockIdx.x, tid = threadIdx.x;
  float tv[4] = {-3e38f, -3e38f, -3e38f, -3e38f};
  int   ti[4] = {0x7fffffff, 0x7fffffff, 0x7fffffff, 0x7fffffff};
  float sh[BEAM]; bool eo[BEAM];
  for (int k = 0; k < BEAM; ++k) {
    int row = b * BEAM + k;
    sh[k] = scores[row] - mx[row] - lse[row];
    eo[k] = (tin[row * LB + t] == EOS_TOK);
  }
  const int total = BEAM * VOC;
  for (int i = tid; i < total; i += 256) {
    int k = i / VOC, vv = i - k * VOC;
    int row = b * BEAM + k;
    float val = eo[k] ? -1e20f : (logits[(size_t)row * VOC + vv] + sh[k]);
    top4_insert(tv, ti, val, i);
  }
#pragma unroll
  for (int k = 0; k < 4; ++k) { lv[tid * 4 + k] = tv[k]; li[tid * 4 + k] = ti[k]; }
  __syncthreads();
  if (tid == 0) {
    float gv[4] = {-3e38f, -3e38f, -3e38f, -3e38f};
    int   gi[4] = {0x7fffffff, 0x7fffffff, 0x7fffffff, 0x7fffffff};
    for (int i = 0; i < 1024; ++i) top4_insert(gv, gi, lv[i], li[i]);
    for (int k = 0; k < BEAM; ++k) { best[b * BEAM + k] = gv[k]; bidx[b * BEAM + k] = gi[k]; }
  }
}

// ---------------- beam state update (tokens gather + finished pool top-k) ----------------
__global__ __launch_bounds__(64)
void k_update(const int* __restrict__ tin, int* __restrict__ tout,
              float* __restrict__ scores, const float* __restrict__ best,
              const int* __restrict__ bidx, float* __restrict__ fin_s,
              int* __restrict__ fin_t, int t) {
  __shared__ int prevK[BEAM], tokn[BEAM];
  const int b = blockIdx.x, tid = threadIdx.x;
  if (tid < BEAM) {
    int idx = bidx[b * BEAM + tid];
    prevK[tid] = idx / VOC;
    tokn[tid]  = idx - (idx / VOC) * VOC;
  }
  __syncthreads();
  for (int i = tid; i < BEAM * LB; i += 64) {
    int k = i / LB, j = i - k * LB;
    int val = tin[(b * BEAM + prevK[k]) * LB + j];
    if (j == t + 1) val = tokn[k];
    tout[(b * BEAM + k) * LB + j] = val;
  }
  if (tid == 0) {
    float pool_s[2 * BEAM]; int pool_t[2 * BEAM][MAXLEN];
    for (int k = 0; k < BEAM; ++k) {
      pool_s[k] = fin_s[b * BEAM + k];
      for (int j = 0; j < MAXLEN; ++j) pool_t[k][j] = fin_t[(b * BEAM + k) * MAXLEN + j];
    }
    for (int k = 0; k < BEAM; ++k) {
      float bs = best[b * BEAM + k];
      scores[b * BEAM + k] = bs;
      pool_s[BEAM + k] = (tokn[k] == EOS_TOK) ? bs : -1e30f;
      for (int j = 0; j < MAXLEN; ++j)   // j < t  ->  new_tokens[k][j+1] == tin[prevK[k]][j+1]
        pool_t[BEAM + k][j] = (j < t) ? tin[(b * BEAM + prevK[k]) * LB + (j + 1)] : 0;
    }
    bool used[2 * BEAM] = {false, false, false, false, false, false, false, false};
    for (int sel = 0; sel < BEAM; ++sel) {
      int bi = -1; float bv = -3.4e38f;
      for (int i = 0; i < 2 * BEAM; ++i)
        if (!used[i] && pool_s[i] > bv) { bv = pool_s[i]; bi = i; }
      used[bi] = true;
      fin_s[b * BEAM + sel] = pool_s[bi];
      for (int j = 0; j < MAXLEN; ++j) fin_t[(b * BEAM + sel) * MAXLEN + j] = pool_t[bi][j];
    }
  }
}

// ---------------- final ranking and output write ----------------
__global__ void k_finalize(const int* __restrict__ tokens, const float* __restrict__ scores,
                           const float* __restrict__ fin_s, const int* __restrict__ fin_t,
                           float* __restrict__ out) {
  const int b = blockIdx.x;
  if (threadIdx.x != 0) return;
  float keys[8], raw[8]; int toks[8][MAXLEN];
  for (int k = 0; k < BEAM; ++k) {
    float fs = fin_s[b * BEAM + k];
    keys[k] = (fs > -1e29f) ? fs + 1e9f : fs;
    raw[k] = fs;
    for (int j = 0; j < MAXLEN; ++j) toks[k][j] = fin_t[(b * BEAM + k) * MAXLEN + j];
  }
  for (int k = 0; k < BEAM; ++k) {
    const int* tk = tokens + (b * BEAM + k) * LB;
    int cnt = 0;
    for (int j = 0; j < MAXLEN; ++j) {
      int g = tk[j + 1];
      cnt += (g == EOS_TOK);
      toks[BEAM + k][j] = (cnt == 0) ? g : 0;
    }
    bool unfin = (tk[MAXLEN] != EOS_TOK);
    float sc = scores[b * BEAM + k];
    keys[BEAM + k] = unfin ? sc : -1e30f;
    raw[BEAM + k] = sc;
  }
  bool used[8] = {false, false, false, false, false, false, false, false};
  for (int sel = 0; sel < BEAM; ++sel) {
    int bi = -1; float bv = -3.4e38f;
    for (int i = 0; i < 8; ++i) if (!used[i] && keys[i] > bv) { bv = keys[i]; bi = i; }
    used[bi] = true;
    for (int j = 0; j < MAXLEN; ++j)
      out[(size_t)(b * BEAM + sel) * MAXLEN + j] = (float)toks[bi][j];
    out[BB * BEAM * MAXLEN + b * BEAM + sel] = raw[bi];
  }
}

// ---------------- host orchestration ----------------
enum {
  IN_IDS = 0, IN_MASK, IN_BEAM, IN_MAXLEN,
  P_WORD_EMB, P_POS_EMB, P_EMB_G, P_EMB_B, P_DENSE_W, P_DENSE_B,
  P_ENC_Q, P_ENC_BQ, P_ENC_K, P_ENC_BK, P_ENC_V, P_ENC_BV, P_ENC_O, P_ENC_BO,
  P_DECS_Q, P_DECS_BQ, P_DECS_K, P_DECS_BK, P_DECS_V, P_DECS_BV, P_DECS_O, P_DECS_BO,
  P_DECC_Q, P_DECC_BQ, P_DECC_K, P_DECC_BK, P_DECC_V, P_DECC_BV, P_DECC_O, P_DECC_BO,
  P_ENC_F1, P_ENC_FB1, P_ENC_F2, P_ENC_FB2, P_ENC_G1, P_ENC_B1, P_ENC_G2, P_ENC_B2,
  P_DEC_F1, P_DEC_FB1, P_DEC_F2, P_DEC_FB2, P_DEC_G1, P_DEC_B1, P_DEC_G2, P_DEC_B2,
  P_DEC_G3, P_DEC_B3
};

extern "C" void kernel_launch(void* const* d_in, const int* in_sizes, int n_in,
                              void* d_out, int out_size, void* d_ws, size_t ws_size,
                              hipStream_t stream) {
  (void)in_sizes; (void)n_in; (void)out_size; (void)ws_size;
  const int* ids  = (const int*)d_in[IN_IDS];
  const int* mask = (const int*)d_in[IN_MASK];
  auto F = [&](int i) { return (const float*)d_in[i]; };

  const float* wemb = F(P_WORD_EMB);
  const float* pemb = F(P_POS_EMB);

  // ---- workspace carving ----
  float* W = (float*)d_ws;
  size_t off = 0;
  auto alloc = [&](size_t n) { float* p = W + off; off += n; return p; };
  float* encbias = alloc(BB * SS);
  float* xe   = alloc((size_t)RENC * H);   // encoder stream / memory
  float* bq   = alloc((size_t)RENC * H);
  float* bk   = alloc((size_t)RENC * H);
  float* bv   = alloc((size_t)RENC * H);
  float* bo   = alloc((size_t)RENC * H);   // attention output
  float* bp   = alloc((size_t)RENC * H);   // projection / residual branch
  float* ffh  = alloc((size_t)RENC * FFD);
  float* memK = alloc((size_t)NDEC * RENC * H);
  float* memV = alloc((size_t)NDEC * RENC * H);
  float* xd   = alloc((size_t)RDEC * H);
  float* hdec = alloc((size_t)RB * H);
  float* hd2  = alloc((size_t)RB * H);
  float* logits = alloc((size_t)RB * VOC);
  float* mxb  = alloc(RB);
  float* lseb = alloc(RB);
  float* bestb   = alloc(RB);
  float* scoresb = alloc(RB);
  float* finsb   = alloc(RB);
  int* tok0  = (int*)alloc(RDEC);
  int* tok1  = (int*)alloc(RDEC);
  int* bidxb = (int*)alloc(RB);
  int* fint  = (int*)alloc(BB * BEAM * MAXLEN);

  auto gemm = [&](const float* A, const float* Bm, const float* bias, float* C,
                  int M, int N, int K, int transB, int act) {
    dim3 g((N + 63) / 64, (M + 31) / 32);
    k_gemm_wmma<<<g, 128, 0, stream>>>(A, Bm, bias, C, M, N, K, transB, act);
  };
  const float scale = 0.125f; // 1/sqrt(64)
  const size_t HH = (size_t)H * H;

  // ================= encoder =================
  k_encbias<<<1, 256, 0, stream>>>(mask, encbias);
  k_embed<<<(RENC * H + 255) / 256, 256, 0, stream>>>(ids, wemb, pemb, xe, RENC, SS);
  k_add_ln<<<RENC, 256, 0, stream>>>(xe, nullptr, F(P_EMB_G), F(P_EMB_B));
  for (int i = 0; i < NENC; ++i) {
    gemm(xe, F(P_ENC_Q) + i * HH, F(P_ENC_BQ) + i * H, bq, RENC, H, H, 0, ACT_NONE);
    gemm(xe, F(P_ENC_K) + i * HH, F(P_ENC_BK) + i * H, bk, RENC, H, H, 0, ACT_NONE);
    gemm(xe, F(P_ENC_V) + i * HH, F(P_ENC_BV) + i * H, bv, RENC, H, H, 0, ACT_NONE);
    k_attn<<<dim3(SS, NHEAD, BB), 256, 0, stream>>>(bq, bk, bv, bo, encbias, 1, 0, SS, SS, 1, scale);
    gemm(bo, F(P_ENC_O) + i * HH, F(P_ENC_BO) + i * H, bp, RENC, H, H, 0, ACT_NONE);
    k_add_ln<<<RENC, 256, 0, stream>>>(xe, bp, F(P_ENC_G1) + i * H, F(P_ENC_B1) + i * H);
    gemm(xe, F(P_ENC_F1) + (size_t)i * H * FFD, F(P_ENC_FB1) + i * FFD, ffh, RENC, FFD, H, 0, ACT_GELU);
    gemm(ffh, F(P_ENC_F2) + (size_t)i * FFD * H, F(P_ENC_FB2) + i * H, bp, RENC, H, FFD, 0, ACT_NONE);
    k_add_ln<<<RENC, 256, 0, stream>>>(xe, bp, F(P_ENC_G2) + i * H, F(P_ENC_B2) + i * H);
  }
  // hoisted cross-attention K/V of encoder memory (step-invariant)
  for (int i = 0; i < NDEC; ++i) {
    gemm(xe, F(P_DECC_K) + i * HH, F(P_DECC_BK) + i * H, memK + (size_t)i * RENC * H, RENC, H, H, 0, ACT_NONE);
    gemm(xe, F(P_DECC_V) + i * HH, F(P_DECC_BV) + i * H, memV + (size_t)i * RENC * H, RENC, H, H, 0, ACT_NONE);
  }

  // ================= beam search =================
  k_init_state<<<1, 128, 0, stream>>>(tok0, scoresb, finsb, fint);
  for (int t = 0; t < MAXLEN; ++t) {
    int* tin  = (t & 1) ? tok1 : tok0;
    int* tout = (t & 1) ? tok0 : tok1;
    k_embed<<<(RDEC * H + 255) / 256, 256, 0, stream>>>(tin, wemb, pemb, xd, RDEC, LB);
    k_add_ln<<<RDEC, 256, 0, stream>>>(xd, nullptr, F(P_EMB_G), F(P_EMB_B));
    for (int i = 0; i < NDEC; ++i) {
      // self attention (causal)
      gemm(xd, F(P_DECS_Q) + i * HH, F(P_DECS_BQ) + i * H, bq, RDEC, H, H, 0, ACT_NONE);
      gemm(xd, F(P_DECS_K) + i * HH, F(P_DECS_BK) + i * H, bk, RDEC, H, H, 0, ACT_NONE);
      gemm(xd, F(P_DECS_V) + i * HH, F(P_DECS_BV) + i * H, bv, RDEC, H, H, 0, ACT_NONE);
      k_attn<<<dim3(LB, NHEAD, RB), 256, 0, stream>>>(bq, bk, bv, bo, nullptr, 1, 1, LB, LB, 1, scale);
      gemm(bo, F(P_DECS_O) + i * HH, F(P_DECS_BO) + i * H, bp, RDEC, H, H, 0, ACT_NONE);
      k_add_ln<<<RDEC, 256, 0, stream>>>(xd, bp, F(P_DEC_G1) + i * H, F(P_DEC_B1) + i * H);
      // cross attention (beam-broadcast memory)
      gemm(xd, F(P_DECC_Q) + i * HH, F(P_DECC_BQ) + i * H, bq, RDEC, H, H, 0, ACT_NONE);
      k_attn<<<dim3(LB, NHEAD, RB), 256, 0, stream>>>(bq, memK + (size_t)i * RENC * H,
                                                      memV + (size_t)i * RENC * H, bo,
                                                      encbias, BEAM, 0, LB, SS, BEAM, scale);
      gemm(bo, F(P_DECC_O) + i * HH, F(P_DECC_BO) + i * H, bp, RDEC, H, H, 0, ACT_NONE);
      k_add_ln<<<RDEC, 256, 0, stream>>>(xd, bp, F(P_DEC_G2) + i * H, F(P_DEC_B2) + i * H);
      // FFN (relu)
      gemm(xd, F(P_DEC_F1) + (size_t)i * H * FFD, F(P_DEC_FB1) + i * FFD, ffh, RDEC, FFD, H, 0, ACT_RELU);
      gemm(ffh, F(P_DEC_F2) + (size_t)i * FFD * H, F(P_DEC_FB2) + i * H, bp, RDEC, H, FFD, 0, ACT_NONE);
      k_add_ln<<<RDEC, 256, 0, stream>>>(xd, bp, F(P_DEC_G3) + i * H, F(P_DEC_B3) + i * H);
    }
    // logits for position t
    k_extract<<<(RB * H + 255) / 256, 256, 0, stream>>>(xd, hdec, t);
    gemm(hdec, F(P_DENSE_W), F(P_DENSE_B), hd2, RB, H, H, 0, ACT_TANH);
    gemm(hd2, wemb, nullptr, logits, RB, VOC, H, 1 /*transB: word_emb^T*/, ACT_NONE);
    k_row_lse<<<RB, 256, 0, stream>>>(logits, mxb, lseb, VOC);
    k_topk<<<BB, 256, 0, stream>>>(logits, mxb, lseb, scoresb, tin, t, bestb, bidxb);
    k_update<<<BB, 64, 0, stream>>>(tin, tout, scoresb, bestb, bidxb, finsb, fint, t);
  }
  // after t = MAXLEN-1 (odd), final tokens live in tok0
  k_finalize<<<BB, 32, 0, stream>>>(tok0, scoresb, finsb, fint, (float*)d_out);
}
